// GroupQueryAttention_22368189677672
// MI455X (gfx1250) — matrix operand order repaired and checked by reading, not compiled, not run on hardware
//
#include <hip/hip_runtime.h>

typedef __attribute__((ext_vector_type(16))) __bf16         v16bf;
typedef __attribute__((ext_vector_type(8)))  float          v8f;
typedef __attribute__((ext_vector_type(16))) unsigned short v16us;
typedef __attribute__((ext_vector_type(8)))  unsigned short v8us;

#define HIDDEN 2048
#define KVDIM  512
#define NHEAD  16
#define HDIM   128
#define SEQ    2048
#define BATCH  2

union frag_cast { v16us u; v16bf b; };

__device__ __forceinline__ unsigned short f2bf(float f) {
  union { float f; unsigned int u; } c; c.f = f;
  unsigned int u = c.u;
  u += 0x7FFFu + ((u >> 16) & 1u);   // round-to-nearest-even
  return (unsigned short)(u >> 16);
}

// LDS byte offset of a flat pointer into __shared__ (low 32 bits of the
// shared-aperture flat address are the LDS offset).
__device__ __forceinline__ unsigned lds_off(const void* p) {
  return (unsigned)(unsigned long long)(uintptr_t)p;
}

// CDNA5 async global->LDS copy, 16 bytes per lane, tracked by ASYNCcnt.
__device__ __forceinline__ void async_load_b128(unsigned lds, const void* gaddr) {
  asm volatile("global_load_async_to_lds_b128 %0, %1, off"
               :: "v"(lds), "v"((unsigned long long)(uintptr_t)gaddr)
               : "memory");
}

// ---------------------------------------------------------------- convert
__global__ __launch_bounds__(256)
void cvt_f32_bf16(const float* __restrict__ in, unsigned short* __restrict__ out, int n) {
  int i = blockIdx.x * blockDim.x + threadIdx.x;
  int stride = gridDim.x * blockDim.x;
  for (; i < n; i += stride) out[i] = f2bf(in[i]);
}

// ---------------------------------------------------------------- GEMM: C[M,N] = A[M,K] * W[N,K]^T + bias
// Block = 128 threads (4 waves). Block tile 64x64: wave w owns rows m0+16w, all 64 cols.
template <bool STORE_BF16>
__global__ __launch_bounds__(128)
void wmma_gemm_nt(const unsigned short* __restrict__ A,
                  const unsigned short* __restrict__ W,
                  const float* __restrict__ bias,
                  void* __restrict__ C,
                  int M, int N, int K) {
  const int wave = threadIdx.x >> 5;
  const int lane = threadIdx.x & 31;
  const int l16  = lane & 15;
  const int hi   = lane >> 4;          // 0 or 1
  const int m0 = blockIdx.x * 64 + wave * 16;
  const int n0 = blockIdx.y * 64;

  v8f acc[4] = {};
  const unsigned short* arow  = A + (size_t)(m0 + l16) * K;
  const unsigned short* wbase = W + (size_t)(n0 + l16) * K;

  for (int k0 = 0; k0 < K; k0 += 32) {
    // A fragment (16x32 bf16): lanes 0-15 -> K {0..7,16..23}; lanes 16-31 -> K {8..15,24..31}
    frag_cast fa;
    {
      const unsigned short* p = arow + k0 + (hi ? 8 : 0);
      v8us a0 = *(const v8us*)(p);
      v8us a1 = *(const v8us*)(p + 16);
#pragma unroll
      for (int i = 0; i < 8; ++i) { fa.u[i] = a0[i]; fa.u[i + 8] = a1[i]; }
    }
#pragma unroll
    for (int t = 0; t < 4; ++t) {
      // B fragment (32x16 bf16): lane n holds column n; lanes 0-15 K {0..7,16..23}, lanes 16-31 K {8..15,24..31}
      frag_cast fb;
      const unsigned short* p = wbase + (size_t)t * 16 * K + k0 + (hi ? 8 : 0);
      v8us b0 = *(const v8us*)(p);
      v8us b1 = *(const v8us*)(p + 16);
#pragma unroll
      for (int i = 0; i < 8; ++i) { fb.u[i] = b0[i]; fb.u[i + 8] = b1[i]; }
      acc[t] = __builtin_amdgcn_wmma_f32_16x16x32_bf16(
          false, fa.b, false, fb.b, (short)0, acc[t], false, false);
    }
  }

#pragma unroll
  for (int t = 0; t < 4; ++t) {
    const int n = n0 + t * 16 + l16;
    const float bvv = bias[n];
#pragma unroll
    for (int v = 0; v < 8; ++v) {
      const int m = m0 + v + (hi ? 8 : 0);
      const float val = acc[t][v] + bvv;
      if (STORE_BF16)
        ((unsigned short*)C)[(size_t)m * N + n] = f2bf(val);
      else
        ((float*)C)[(size_t)m * N + n] = val;
    }
  }
}

// ---------------------------------------------------------------- attention
// grid = B*H*(S/16) blocks, 128 threads (4 waves).
// LDS: scores[16][2048] f32 (128KB) + reductions + double-buffered V tile (16KB).
__global__ __launch_bounds__(128)
void gqa_attention(const unsigned short* __restrict__ Q,   // [B*S, 2048] bf16
                   const unsigned short* __restrict__ Km,  // [B*S, 512]  bf16
                   const unsigned short* __restrict__ Vm,  // [B*S, 512]  bf16
                   unsigned short* __restrict__ AO) {      // [B*2048, 2048] bf16
  extern __shared__ char smem[];
  float* sc             = (float*)smem;                      // 16*2048 f32
  float* redm           = (float*)(smem + 16 * SEQ * 4);     // [16][8]
  float* reds           = redm + 128;                        // [16][8]
  float* rowinv         = reds + 128;                        // [16]
  unsigned short* vtile = (unsigned short*)(rowinv + 16);    // [2][32][128] bf16

  const int bid = blockIdx.x;
  const int s0  = (bid & 127) * 16;
  const int h   = (bid >> 7) & 15;
  const int b   = bid >> 11;
  const int kvh = h >> 2;

  const int wave = threadIdx.x >> 5;
  const int lane = threadIdx.x & 31;
  const int l16  = lane & 15;
  const int hi   = lane >> 4;

  const unsigned short* qbase = Q  + (size_t)b * SEQ * HIDDEN + (size_t)h   * HDIM;
  const unsigned short* kbase = Km + (size_t)b * SEQ * KVDIM  + (size_t)kvh * HDIM;
  const unsigned short* vbase = Vm + (size_t)b * SEQ * KVDIM  + (size_t)kvh * HDIM;

  // Q tile as 4 A-fragments (d = 0..127 in chunks of 32)
  v16bf aq[4];
  {
    const unsigned short* qrow = qbase + (size_t)(s0 + l16) * HIDDEN;
#pragma unroll
    for (int c = 0; c < 4; ++c) {
      frag_cast fa;
      const unsigned short* p = qrow + c * 32 + (hi ? 8 : 0);
      v8us a0 = *(const v8us*)(p);
      v8us a1 = *(const v8us*)(p + 16);
#pragma unroll
      for (int i = 0; i < 8; ++i) { fa.u[i] = a0[i]; fa.u[i + 8] = a1[i]; }
      aq[c] = fa.b;
    }
  }

  const float scale = 0.08838834764831845f;  // 1/sqrt(128)

  // ---- scores: wave covers keys [wave*512, wave*512+512)
  for (int blk = 0; blk < 32; ++blk) {
    const int kb = wave * 512 + blk * 16;
    v8f c = {};
#pragma unroll
    for (int cc = 0; cc < 4; ++cc) {
      frag_cast fb;
      const unsigned short* p = kbase + (size_t)(kb + l16) * KVDIM + cc * 32 + (hi ? 8 : 0);
      v8us b0 = *(const v8us*)(p);
      v8us b1 = *(const v8us*)(p + 16);
#pragma unroll
      for (int i = 0; i < 8; ++i) { fb.u[i] = b0[i]; fb.u[i + 8] = b1[i]; }
      c = __builtin_amdgcn_wmma_f32_16x16x32_bf16(
          false, aq[cc], false, fb.b, (short)0, c, false, false);
    }
#pragma unroll
    for (int v = 0; v < 8; ++v) {
      const int r = v + hi * 8;
      sc[r * SEQ + kb + l16] = c[v] * scale;
    }
  }
  __syncthreads();

  // ---- softmax across 2048 keys per row (16 rows, 8 chunks of 256 per row)
  {
    const int r  = threadIdx.x & 15;
    const int ch = threadIdx.x >> 4;
    float* row = sc + r * SEQ + ch * 256;
    float m = -3.0e38f;
    for (int i = 0; i < 256; ++i) m = fmaxf(m, row[i]);
    redm[r * 8 + ch] = m;
    __syncthreads();
    float rm = redm[r * 8];
#pragma unroll
    for (int i = 1; i < 8; ++i) rm = fmaxf(rm, redm[r * 8 + i]);
    float s = 0.f;
    for (int i = 0; i < 256; ++i) {
      float e = __expf(row[i] - rm);
      row[i] = e;
      s += e;
    }
    reds[r * 8 + ch] = s;
    __syncthreads();
    if (threadIdx.x < 16) {
      float t = 0.f;
#pragma unroll
      for (int i = 0; i < 8; ++i) t += reds[threadIdx.x * 8 + i];
      rowinv[threadIdx.x] = 1.0f / t;
    }
    __syncthreads();
  }

  // ---- O = P @ V ; wave owns d tiles [wave*32, wave*32+32)
  // V tiles staged with async global->LDS copies (ASYNCcnt), double-buffered so
  // tile i+1's HBM/L2 latency overlaps tile i's WMMAs.
  v8f o0 = {}, o1 = {};
  const int d0 = wave * 32;
  const float rinv = rowinv[l16];

  const int sr  = threadIdx.x >> 2;   // staging row 0..31
  const int sq4 = threadIdx.x & 3;    // staging quarter (32 cols = 64B each)

  // prologue: stage tile 0 into buffer 0 (4 async b128 per thread = 4 per wave)
  {
    const unsigned short* vrow = vbase + (size_t)sr * KVDIM + sq4 * 32;
    unsigned short* dst = vtile + sr * 128 + sq4 * 32;
#pragma unroll
    for (int c = 0; c < 4; ++c)
      async_load_b128(lds_off(dst + c * 8), vrow + c * 8);
  }

  for (int it = 0; it < 64; ++it) {
    const int cur = it & 1;
    const unsigned short* vt = vtile + cur * (32 * 128);

    if (it < 63) {
      // issue next tile into the other buffer, then wait for current tile only
      const unsigned short* vrow = vbase + (size_t)((it + 1) * 32 + sr) * KVDIM + sq4 * 32;
      unsigned short* dst = vtile + (1 - cur) * (32 * 128) + sr * 128 + sq4 * 32;
#pragma unroll
      for (int c = 0; c < 4; ++c)
        async_load_b128(lds_off(dst + c * 8), vrow + c * 8);
      asm volatile("s_wait_asynccnt 0x4" ::: "memory");   // current tile landed
    } else {
      asm volatile("s_wait_asynccnt 0x0" ::: "memory");
    }
    __syncthreads();   // all waves' current-tile writes visible

    // P fragment from LDS scores (normalize while converting to bf16)
    frag_cast fp;
    {
      const int kc = it * 32;
      const float* prow = sc + l16 * SEQ + kc + (hi ? 16 : 0);
#pragma unroll
      for (int i = 0; i < 8; ++i) fp.u[i]     = f2bf(prow[i] * rinv);
#pragma unroll
      for (int i = 0; i < 8; ++i) fp.u[i + 8] = f2bf(prow[8 + i] * rinv);
    }

    // two B fragments from staged V
    frag_cast fb0, fb1;
    {
      const int koff = hi ? 16 : 0;
#pragma unroll
      for (int i = 0; i < 16; ++i) {
        fb0.u[i] = vt[(koff + i) * 128 + d0 + l16];
        fb1.u[i] = vt[(koff + i) * 128 + d0 + 16 + l16];
      }
    }
    o0 = __builtin_amdgcn_wmma_f32_16x16x32_bf16(false, fp.b, false, fb0.b, (short)0, o0, false, false);
    o1 = __builtin_amdgcn_wmma_f32_16x16x32_bf16(false, fp.b, false, fb1.b, (short)0, o1, false, false);
    __syncthreads();   // everyone done reading `cur` before it is overwritten
  }

  // store in the reference's reshape-without-transpose layout:
  // (b,h,s,d) -> row b*2048 + h*128 + s0/16 , col (s&15)*128 + d
  unsigned short* aorow = AO + ((size_t)b * SEQ + (size_t)h * HDIM + (s0 >> 4)) * HIDDEN;
#pragma unroll
  for (int v = 0; v < 8; ++v) {
    const int srow = v + hi * 8;
    aorow[srow * HDIM + d0 + l16]      = f2bf(o0[v]);
    aorow[srow * HDIM + d0 + 16 + l16] = f2bf(o1[v]);
  }
}

// ---------------------------------------------------------------- launcher
extern "C" void kernel_launch(void* const* d_in, const int* in_sizes, int n_in,
                              void* d_out, int out_size, void* d_ws, size_t ws_size,
                              hipStream_t stream) {
  const float* x  = (const float*)d_in[0];
  const float* wq = (const float*)d_in[1];
  const float* bq = (const float*)d_in[2];
  const float* wk = (const float*)d_in[3];
  const float* bk = (const float*)d_in[4];
  const float* wv = (const float*)d_in[5];
  const float* bv = (const float*)d_in[6];
  const float* wo = (const float*)d_in[7];
  const float* bo = (const float*)d_in[8];
  float* out = (float*)d_out;

  const size_t ROWS = (size_t)BATCH * SEQ;  // 4096
  char* ws = (char*)d_ws;
  size_t off = 0;
  unsigned short* xb  = (unsigned short*)(ws + off); off += ROWS * HIDDEN * 2;           // 16 MB
  unsigned short* wqb = (unsigned short*)(ws + off); off += (size_t)HIDDEN * HIDDEN * 2; // 8 MB
  unsigned short* wkb = (unsigned short*)(ws + off); off += (size_t)KVDIM * HIDDEN * 2;  // 2 MB
  unsigned short* wvb = (unsigned short*)(ws + off); off += (size_t)KVDIM * HIDDEN * 2;  // 2 MB
  unsigned short* wob = (unsigned short*)(ws + off); off += (size_t)HIDDEN * HIDDEN * 2; // 8 MB
  unsigned short* qb  = (unsigned short*)(ws + off); off += ROWS * HIDDEN * 2;           // 16 MB
  unsigned short* kb  = (unsigned short*)(ws + off); off += ROWS * KVDIM * 2;            // 4 MB
  unsigned short* vb  = (unsigned short*)(ws + off); off += ROWS * KVDIM * 2;            // 4 MB
  unsigned short* aob = (unsigned short*)(ws + off); off += ROWS * HIDDEN * 2;           // 16 MB

  // 1) convert inputs/weights to bf16
  cvt_f32_bf16<<<1024, 256, 0, stream>>>(x,  xb,  (int)(ROWS * HIDDEN));
  cvt_f32_bf16<<<1024, 256, 0, stream>>>(wq, wqb, HIDDEN * HIDDEN);
  cvt_f32_bf16<<<512,  256, 0, stream>>>(wk, wkb, KVDIM * HIDDEN);
  cvt_f32_bf16<<<512,  256, 0, stream>>>(wv, wvb, KVDIM * HIDDEN);
  cvt_f32_bf16<<<1024, 256, 0, stream>>>(wo, wob, HIDDEN * HIDDEN);

  // 2) Q/K/V projections (store bf16)
  wmma_gemm_nt<true><<<dim3(64, 32), 128, 0, stream>>>(xb, wqb, bq, qb, (int)ROWS, HIDDEN, HIDDEN);
  wmma_gemm_nt<true><<<dim3(64, 8),  128, 0, stream>>>(xb, wkb, bk, kb, (int)ROWS, KVDIM,  HIDDEN);
  wmma_gemm_nt<true><<<dim3(64, 8),  128, 0, stream>>>(xb, wvb, bv, vb, (int)ROWS, KVDIM,  HIDDEN);

  // 3) attention
  const size_t smem = (size_t)16 * SEQ * 4            // scores
                    + 128 * 4 + 128 * 4 + 16 * 4      // reductions
                    + (size_t)2 * 32 * 128 * 2;       // double-buffered V tile
  gqa_attention<<<BATCH * NHEAD * (SEQ / 16), 128, smem, stream>>>(qb, kb, vb, aob);

  // 4) output projection (store f32)
  wmma_gemm_nt<false><<<dim3(64, 32), 128, 0, stream>>>(aob, wob, bo, out, (int)ROWS, HIDDEN, HIDDEN);
}
